// ContGen_83399674953736
// MI455X (gfx1250) — matrix operand order repaired and checked by reading, not compiled, not run on hardware
//
#include <hip/hip_runtime.h>
#include <math.h>

typedef __attribute__((ext_vector_type(16))) _Float16 v16h;
typedef __attribute__((ext_vector_type(8)))  _Float16 v8h;
typedef __attribute__((ext_vector_type(8)))  float    v8f;

#define ACT_RELU 0
#define ACT_TANH 1

// ---------------------------------------------------------------------------
// Weight pre-pack: src w is torch ConvTranspose2d layout (Cin, Cout, 3, 3) f32.
// Tap t=(a,b) of the equivalent conv uses w[ci][co][2-a][2-b].  We emit f16 in
// the per-lane order this code uses for the B (32x16) WMMA operand (contiguous k):
//   dst[((t*KC + q)*NT + j)*512 + lane*16 + e],  k = q*32 + (lane>=16)*16 + e,
//   co = j*16 + (lane&15)   ->  each lane's 16 B elements are contiguous 32 B.
// ---------------------------------------------------------------------------
__global__ void prep_w_kernel(const float* __restrict__ w, _Float16* __restrict__ dst,
                              int Cin, int Cout, int NT, int total) {
    int idx = blockIdx.x * 256 + threadIdx.x;
    if (idx >= total) return;
    int KC   = Cin >> 5;
    int e    = idx & 15;
    int lane = (idx >> 4) & 31;
    int rest = idx >> 9;
    int j = rest % NT; rest /= NT;
    int q = rest % KC;
    int t = rest / KC;
    int a = t / 3, b = t - 3 * a;
    int k  = (q << 5) + ((lane >> 4) << 4) + e;
    int co = (j << 4) + (lane & 15);
    float v = 0.f;
    if (co < Cout)
        v = w[((size_t)(k * Cout + co) * 3 + (2 - a)) * 3 + (2 - b)];
    dst[idx] = (_Float16)v;
}

__device__ __forceinline__ float apply_act(float v, int act) {
    return (act == ACT_TANH) ? tanhf(v) : fmaxf(v, 0.f);
}

// ---------------------------------------------------------------------------
// ConvTranspose2d(k=3, pad=1, [stride 2 => output_padding 1]) as implicit GEMM
// on v_wmma_f32_16x16x32_f16.  One wave: 16 output pixels x 2x16 output chans.
// LDS patch sA[a][u][k] (k innermost, padded 32->40 halfs = 80 B row stride):
//   value feeding vertical tap a at output-x slot u (output coord ox0-1+u).
// For stride 2 parity zeros implement the input dilation, so the per-tap
// gather is identical for both strides.  A fragment = 2x ds_load_b128.
// ---------------------------------------------------------------------------
template<int STRIDE>
__global__ __launch_bounds__(128)
void convt_wmma_kernel(const float* __restrict__ in, const _Float16* __restrict__ wpk,
                       float* __restrict__ out,
                       int Cin, int Cout, int Hin, int Win, int Hout, int Wout, int act) {
    __shared__ __align__(128) _Float16 sA[3][18][40];

    const int tid     = threadIdx.x;
    const int wave    = tid >> 5;
    const int lane    = tid & 31;
    const int m       = lane & 15;      // output pixel within tile
    const int halfSel = lane >> 4;

    const int tilesX = Wout >> 4;
    const int oy     = blockIdx.x / tilesX;
    const int ox0    = (blockIdx.x - oy * tilesX) << 4;
    const int batch  = blockIdx.z;
    const int NT     = (Cout + 15) >> 4;
    const int j0     = (blockIdx.y * 4 + wave) * 2;  // two cout tiles per wave
    const int jj0    = (j0     < NT) ? j0     : NT - 1;
    const int jj1    = (j0 + 1 < NT) ? j0 + 1 : NT - 1;
    const int KC     = Cin >> 5;

    const float* inB = in + (size_t)batch * Cin * Hin * Win;
    const size_t chanStride = (size_t)Hin * Win;

    v8f acc0 = {}, acc1 = {};
    for (int q = 0; q < KC; ++q) {
        // prefetch one row of the next cin-chunk (global_prefetch_b8)
        if (q + 1 < KC) {
            int pk = tid & 31;
            int py = oy >> (STRIDE - 1);
            py = py < 0 ? 0 : (py >= Hin ? Hin - 1 : py);
            __builtin_prefetch(inB + ((size_t)((q + 1) * 32 + pk)) * chanStride +
                                   (size_t)py * Win, 0, 2);
        }
        __syncthreads();
        // cooperative patch load: 32 cin x 3 vtaps x 18 slots, f32 -> f16.
        // iteration order keeps x (u) fastest -> coalesced global reads.
        for (int i = tid; i < 32 * 3 * 18; i += 128) {
            int k   = i / 54;
            int rem = i - k * 54;
            int a   = rem / 18;
            int u   = rem - a * 18;
            int yo  = oy + a - 1;       // output-space vertical coord
            int xo  = ox0 + u - 1;      // output-space horizontal coord
            float v = 0.f;
            if (STRIDE == 1) {
                if (yo >= 0 && yo < Hin && xo >= 0 && xo < Win)
                    v = inB[((size_t)((q << 5) + k)) * chanStride + (size_t)yo * Win + xo];
            } else {
                if (yo >= 0 && xo >= 0 && !(yo & 1) && !(xo & 1)) {
                    int iy = yo >> 1, ix = xo >> 1;
                    if (iy < Hin && ix < Win)
                        v = inB[((size_t)((q << 5) + k)) * chanStride + (size_t)iy * Win + ix];
                }
            }
            sA[a][u][k] = (_Float16)v;
        }
        __syncthreads();

#pragma unroll
        for (int t = 0; t < 9; ++t) {
            const int a = t / 3, b = t - 3 * a;
            // A fragment: two contiguous 16B LDS chunks per lane
            const _Float16* rowp = &sA[a][m + b][halfSel << 4];
            v8h lo = *(const v8h*)rowp;
            v8h hi = *(const v8h*)(rowp + 8);
            v16h aF = __builtin_shufflevector(lo, hi, 0, 1, 2, 3, 4, 5, 6, 7,
                                              8, 9, 10, 11, 12, 13, 14, 15);
            // B fragments: contiguous 32B per lane from pre-packed weights
            const size_t tb = ((size_t)t * KC + q) * NT;
            v16h bF0 = *(const v16h*)(wpk + ((tb + jj0) << 9) + ((size_t)lane << 4));
            v16h bF1 = *(const v16h*)(wpk + ((tb + jj1) << 9) + ((size_t)lane << 4));
            acc0 = __builtin_amdgcn_wmma_f32_16x16x32_f16(
                false, aF, false, bF0, (short)0, acc0, false, false);
            acc1 = __builtin_amdgcn_wmma_f32_16x16x32_f16(
                false, aF, false, bF1, (short)0, acc1, false, false);
        }
    }

    const int n = lane & 15;
#pragma unroll
    for (int s = 0; s < 2; ++s) {
        int j  = j0 + s;
        v8f* ap = s ? &acc1 : &acc0;
        int co = (j << 4) + n;
        if (j < NT && co < Cout) {
            float* ob = out + (((size_t)batch * Cout + co) * Hout + oy) * Wout + ox0 +
                        (halfSel << 3);
            v8f a = *ap;
            float4 s0 = make_float4(apply_act(a[0], act), apply_act(a[1], act),
                                    apply_act(a[2], act), apply_act(a[3], act));
            float4 s1 = make_float4(apply_act(a[4], act), apply_act(a[5], act),
                                    apply_act(a[6], act), apply_act(a[7], act));
            *(float4*)(ob)     = s0;
            *(float4*)(ob + 4) = s1;
        }
    }
}

// ---------------------------------------------------------------------------
// Adaptive separable conv (K=5, replicate pad) + mask blend.  Bandwidth bound.
// ---------------------------------------------------------------------------
__global__ void fusion_kernel(const float* __restrict__ feat, const float* __restrict__ kv,
                              const float* __restrict__ kh, const float* __restrict__ mask,
                              float* __restrict__ outp, int C, int H, int W, int total) {
    int idx = blockIdx.x * 256 + threadIdx.x;
    if (idx >= total) return;
    int x  = idx % W;
    int y  = (idx / W) % H;
    int c  = (idx / (W * H)) % C;
    int bI = idx / (W * H * C);
    const int HW = H * W;

    const float* kvp = kv + ((size_t)bI * 5) * HW + y * W + x;
    const float* khp = kh + ((size_t)bI * 5) * HW + y * W + x;
    float kvv[5], khv[5];
#pragma unroll
    for (int i = 0; i < 5; ++i) { kvv[i] = kvp[(size_t)i * HW]; khv[i] = khp[(size_t)i * HW]; }

    const float* fb = feat + ((size_t)bI * C + c) * HW;
    float acc = 0.f;
#pragma unroll
    for (int i = 0; i < 5; ++i) {
        int yy = y + i - 2; yy = yy < 0 ? 0 : (yy >= H ? H - 1 : yy);
        const float* row = fb + (size_t)yy * W;
#pragma unroll
        for (int jt = 0; jt < 5; ++jt) {
            int xx = x + jt - 2; xx = xx < 0 ? 0 : (xx >= W ? W - 1 : xx);
            acc += row[xx] * kvv[i] * khv[jt];
        }
    }
    float mv = mask[(size_t)bI * HW + y * W + x];
    outp[idx] = mv * acc + (1.f - mv) * fb[(size_t)y * W + x];
}

// ---------------------------------------------------------------------------
extern "C" void kernel_launch(void* const* d_in, const int* in_sizes, int n_in,
                              void* d_out, int out_size, void* d_ws, size_t ws_size,
                              hipStream_t stream) {
    (void)in_sizes; (void)n_in; (void)out_size; (void)ws_size;

    const float* cont = (const float*)d_in[0];
    const float* W1 = (const float*)d_in[1];
    const float* W2 = (const float*)d_in[2];
    const float* W3 = (const float*)d_in[3];
    const float* W4 = (const float*)d_in[4];
    const float* W5 = (const float*)d_in[5];
    const float* W6 = (const float*)d_in[6];
    const float* KV[4] = {(const float*)d_in[7],  (const float*)d_in[10],
                          (const float*)d_in[13], (const float*)d_in[16]};
    const float* KH[4] = {(const float*)d_in[8],  (const float*)d_in[11],
                          (const float*)d_in[14], (const float*)d_in[17]};
    const float* MM[4] = {(const float*)d_in[9],  (const float*)d_in[12],
                          (const float*)d_in[15], (const float*)d_in[18]};

    // output tuple (out, f8, f16, f32) concatenated flat
    float* outT = (float*)d_out;                 // [32,3,128,128]
    float* f8   = outT + 1572864;                // [32,64,16,16]
    float* f16  = outT + 2097152;                // [32,512,32,32]
    float* f32  = outT + 18874368;               // [32,128,64,64]

    char* ws = (char*)d_ws;                      // all offsets 256B aligned
    _Float16* wpk1 = (_Float16*)(ws + 0);         //  9*64*512  f16
    _Float16* wpk2 = (_Float16*)(ws + 589824);    //  9*512*256
    _Float16* wpk3 = (_Float16*)(ws + 2949120);   //  9*256*128
    _Float16* wpk4 = (_Float16*)(ws + 3538944);   //  9*128*128
    _Float16* wpk5 = (_Float16*)(ws + 3833856);   //  9*128*64
    _Float16* wpk6 = (_Float16*)(ws + 3981312);   //  9*64*16
    float* bufA = (float*)(ws + 3999744);         // 32*512*32*32  f32 (67MB)
    float* bufB = (float*)(ws + 71108608);        // 32*256*32*32  f32 (33.5MB)
    float* bufC = (float*)(ws + 104663040);       // 32*128*64*64 / 32*64*128*128 (67MB)
    float* bufD = (float*)(ws + 171771904);       // 32*3*128*128 (6.3MB)

    auto prep = [&](const float* w, _Float16* dst, int Cin, int Cout) {
        int NT = (Cout + 15) / 16;
        int KC = Cin / 32;
        int total = 9 * KC * NT * 512;
        prep_w_kernel<<<(total + 255) / 256, 256, 0, stream>>>(w, dst, Cin, Cout, NT, total);
    };
    prep(W1, wpk1, 64, 512);
    prep(W2, wpk2, 512, 256);
    prep(W3, wpk3, 256, 128);
    prep(W4, wpk4, 128, 128);
    prep(W5, wpk5, 128, 64);
    prep(W6, wpk6, 64, 3);

    auto conv = [&](int stride, const float* in, const _Float16* wp, float* o,
                    int Cin, int Cout, int Hin, int Win, int Hout, int Wout, int act) {
        int NT = (Cout + 15) / 16;
        dim3 grid((Wout / 16) * Hout, (NT + 7) / 8, 32);
        if (stride == 1)
            convt_wmma_kernel<1><<<grid, 128, 0, stream>>>(in, wp, o, Cin, Cout, Hin, Win, Hout, Wout, act);
        else
            convt_wmma_kernel<2><<<grid, 128, 0, stream>>>(in, wp, o, Cin, Cout, Hin, Win, Hout, Wout, act);
    };
    auto fus = [&](const float* feat, int li, float* o, int C, int H) {
        int total = 32 * C * H * H;
        fusion_kernel<<<(total + 255) / 256, 256, 0, stream>>>(feat, KV[li], KH[li], MM[li],
                                                               o, C, H, H, total);
    };

    fus(cont, 0, f8, 64, 16);                                   // f8
    conv(2, f8,   wpk1, bufA, 64, 512, 16, 16, 32, 32, ACT_RELU);
    fus(bufA, 1, f16, 512, 32);                                 // f16
    conv(1, f16,  wpk2, bufB, 512, 256, 32, 32, 32, 32, ACT_RELU);
    conv(2, bufB, wpk3, bufC, 256, 128, 32, 32, 64, 64, ACT_RELU);
    fus(bufC, 2, f32, 128, 64);                                 // f32
    conv(1, f32,  wpk4, bufA, 128, 128, 64, 64, 64, 64, ACT_RELU);   // reuse bufA
    conv(2, bufA, wpk5, bufC, 128, 64, 64, 64, 128, 128, ACT_RELU);  // reuse bufC
    conv(1, bufC, wpk6, bufD, 64, 3, 128, 128, 128, 128, ACT_TANH);
    fus(bufD, 3, outT, 3, 128);                                 // out
}
